// EdgeConv_18657337934215
// MI455X (gfx1250) — matrix operand order repaired and checked by reading, not compiled, not run on hardware
//
#include <hip/hip_runtime.h>
#include <hip/hip_bf16.h>

// EdgeConv for MI455X (gfx1250, wave32).
// f16 operands + f32 accumulation via v_wmma_f32_16x16x32_f16 for both the
// KNN gram matrix and the EdgeConv GEMM. Batches processed sequentially on
// `stream` so one 64MB dist buffer is reused; h kept in f32 (268MB) because
// BatchNorm stats are global across (B,N,K). W is staged into LDS by the
// Tensor Data Mover (tensor_load_to_lds, 6-arg toolchain form) with hardware
// row padding.

typedef __attribute__((ext_vector_type(16))) _Float16 v16h;
typedef __attribute__((ext_vector_type(8)))  _Float16 v8h;
typedef __attribute__((ext_vector_type(8)))  float    v8f;
typedef __attribute__((ext_vector_type(4)))  unsigned int u32x4;
typedef __attribute__((ext_vector_type(8)))  int i32x8;
typedef __attribute__((ext_vector_type(4)))  int i32x4;

#define BATCH 8
#define FDIM 64
#define NPTS 4096
#define KNN 16
#define CIN 128
#define COUT 128
#define NEG_SLOPE 0.2f
#define BN_EPS 1e-5f
#define NTOT (BATCH * NPTS * KNN)  // 524288
#define WLDS_STRIDE 144            // f16 elems: 128 data + 8-dword pad

static __device__ __forceinline__ v16h cat8(v8h lo, v8h hi) {
  return __builtin_shufflevector(lo, hi, 0, 1, 2, 3, 4, 5, 6, 7, 8, 9, 10, 11, 12, 13, 14, 15);
}
static __device__ __forceinline__ v8f wmma_f16(v16h a, v16h b, v8f c) {
  // (neg_a, A, neg_b, B, c_mod, C, reuse_a, reuse_b)
  return __builtin_amdgcn_wmma_f32_16x16x32_f16(false, a, false, b, (short)0, c, false, false);
}

// ---- Stage 0a: x (B,F,N) f32 -> xt (B,N,F) f16 ; xx[b][n] = sum_f x^2 ------
__global__ void k_prep_x(const float* __restrict__ x, _Float16* __restrict__ xt,
                         float* __restrict__ xx) {
  int n = blockIdx.x * blockDim.x + threadIdx.x;
  int b = blockIdx.y;
  const float* src = x + (size_t)b * FDIM * NPTS + n;
  _Float16* dst = xt + ((size_t)b * NPTS + n) * FDIM;
  float acc = 0.f;
#pragma unroll
  for (int f = 0; f < FDIM; ++f) {
    float v = src[(size_t)f * NPTS];
    acc += v * v;
    dst[f] = (_Float16)v;
  }
  xx[b * NPTS + n] = acc;
}

// ---- Stage 0b: W f32 -> f16, zero global BN accumulators -------------------
__global__ void k_prep_w(const float* __restrict__ W, _Float16* __restrict__ Wh,
                         float* __restrict__ gSum, float* __restrict__ gSq) {
  int i = blockIdx.x * blockDim.x + threadIdx.x;
  if (i < COUT * CIN) Wh[i] = (_Float16)W[i];
  if (blockIdx.x == 0 && threadIdx.x < COUT) {
    gSum[threadIdx.x] = 0.f;
    gSq[threadIdx.x] = 0.f;
  }
}

// ---- Stage 1: dist[n][m] = 2*<x_n,x_m> - |x_n|^2 - |x_m|^2 (one batch) -----
// One wave computes a 16x16 tile with two v_wmma_f32_16x16x32_f16 (K=64).
__global__ void __launch_bounds__(256) k_gram(const _Float16* __restrict__ xt,
                                              const float* __restrict__ xx,
                                              float* __restrict__ dist, int b) {
  const int lane = threadIdx.x & 31;
  const int wave = threadIdx.x >> 5;
  const int l16 = lane & 15;
  const int hi = lane >> 4;
  const int n0 = blockIdx.x * 16;
  const int m0 = (blockIdx.y * 8 + wave) * 16;
  const _Float16* xb = xt + (size_t)b * NPTS * FDIM;
  const float* xxb = xx + b * NPTS;

  // A fragment: row = n0 + lane%16, K-chunks [hi*8,+8) and [16+hi*8,+8)
  const _Float16* arow = xb + (size_t)(n0 + l16) * FDIM;
  v16h A0 = cat8(*(const v8h*)(arow + hi * 8), *(const v8h*)(arow + 16 + hi * 8));
  v16h A1 = cat8(*(const v8h*)(arow + 32 + hi * 8), *(const v8h*)(arow + 48 + hi * 8));

  // B fragment: col = m0 + lane%16, K-chunks [hi*8,+8) and [16+hi*8,+8)
  const _Float16* brow = xb + (size_t)(m0 + l16) * FDIM;
  v16h B0 = cat8(*(const v8h*)(brow + hi * 8), *(const v8h*)(brow + 16 + hi * 8));
  v16h B1 = cat8(*(const v8h*)(brow + 32 + hi * 8), *(const v8h*)(brow + 48 + hi * 8));

  v8f acc = {};
  acc = wmma_f16(A0, B0, acc);
  acc = wmma_f16(A1, B1, acc);

  float xxm = xxb[m0 + l16];
#pragma unroll
  for (int r = 0; r < 8; ++r) {
    int M = r + hi * 8;  // C layout: lanes0-15 -> M=r, lanes16-31 -> M=r+8
    float d = 2.0f * acc[r] - xxb[n0 + M] - xxm;
    dist[(size_t)(n0 + M) * NPTS + (m0 + l16)] = d;
  }
}

// ---- Stage 2: top-16 per row; one wave per row, register top-k + shuffle ---
__global__ void __launch_bounds__(256) k_topk(const float* __restrict__ dist,
                                              int* __restrict__ idx, int b) {
  const int lane = threadIdx.x & 31;
  const int n = blockIdx.x * 8 + (threadIdx.x >> 5);
  const float* drow = dist + (size_t)n * NPTS;

  float tv[16];
  int ti[16];
#pragma unroll
  for (int j = 0; j < 16; ++j) { tv[j] = -3.0e38f; ti[j] = 0; }
  float minv = -3.0e38f;

  for (int c = lane; c < NPTS; c += 32) {
    float v = drow[c];
    if (v > minv) {
      int am = 0;
      float mv = tv[0];
#pragma unroll
      for (int j = 1; j < 16; ++j)
        if (tv[j] < mv) { mv = tv[j]; am = j; }
#pragma unroll
      for (int j = 0; j < 16; ++j)
        if (j == am) { tv[j] = v; ti[j] = c; }
      mv = tv[0];
#pragma unroll
      for (int j = 1; j < 16; ++j) mv = fminf(mv, tv[j]);
      minv = mv;
    }
  }

  int* iout = idx + ((size_t)b * NPTS + n) * KNN;
  // 16 rounds of wave-wide argmax over the 32x16 register pool
  for (int r = 0; r < KNN; ++r) {
    float bv = tv[0];
    int bs = 0;
#pragma unroll
    for (int j = 1; j < 16; ++j)
      if (tv[j] > bv) { bv = tv[j]; bs = j; }
    int payload = (lane << 8) | bs;
#pragma unroll
    for (int off = 16; off > 0; off >>= 1) {
      float ov = __shfl_xor(bv, off, 32);
      int op = __shfl_xor(payload, off, 32);
      if (ov > bv || (ov == bv && op < payload)) { bv = ov; payload = op; }
    }
    int wl = payload >> 8, wsl = payload & 255;
    if (lane == wl) {
      int sel = ti[0];
#pragma unroll
      for (int j = 1; j < 16; ++j)
        if (j == wsl) sel = ti[j];
      iout[r] = sel;
#pragma unroll
      for (int j = 0; j < 16; ++j)
        if (j == wsl) tv[j] = -3.0e38f;
    }
  }
}

// ---- Stage 3: h = feat @ W^T, + BN partial sums. One wave per point n. -----
// 16 neighbors of n = exactly one 16-row M tile; K=128 -> 4 WMMA k-steps.
// W (128x128 f16) staged global->LDS by the Tensor Data Mover with hardware
// padding: 64-dword rows + 8-dword pad -> LDS row stride 144 f16 (bank spread).
__global__ void __launch_bounds__(256) k_edgeconv(
    const _Float16* __restrict__ xt, const _Float16* __restrict__ Wh,
    const int* __restrict__ idx, float* __restrict__ h,
    float* __restrict__ gSum, float* __restrict__ gSq) {
  __shared__ _Float16 Wl[COUT * WLDS_STRIDE];
  __shared__ float sSum[COUT], sSq[COUT];

#if __has_builtin(__builtin_amdgcn_tensor_load_to_lds) && __has_builtin(__builtin_amdgcn_s_wait_tensorcnt)
  if (threadIdx.x < 32) {  // one wave issues the TDM DMA for the block
    unsigned lds_base = (unsigned)(unsigned long long)(void*)&Wl[0];  // addr[31:0] = LDS offset
    unsigned long long ga = (unsigned long long)(const void*)Wh;
    u32x4 g0;
    g0[0] = 1u;                                   // count=1, user descriptor
    g0[1] = lds_base;                             // lds_addr [63:32]
    g0[2] = (unsigned)(ga & 0xffffffffu);         // global_addr [95:64]
    g0[3] = (unsigned)((ga >> 32) & 0x01ffffffu)  // global_addr [120:96]
            | (2u << 30);                         // type=2 ("image")
    i32x8 g1;
    g1[0] = (int)((1u << 16)      // data_size = 2 bytes
                  | (1u << 20)    // pad_enable
                  | (5u << 22)    // pad_interval: 64 dwords (256B row)
                  | (7u << 25));  // pad_amount: 8 dwords (32B)
    g1[1] = (int)(128u << 16);    // tensor_dim0[15:0] @ bits 63:48
    g1[2] = (int)(128u << 16);    // tensor_dim1[15:0] @ bits 95:80
    g1[3] = (int)(128u << 16);    // tile_dim0 @ bits 127:112
    g1[4] = 128;                  // tile_dim1 @ bits 143:128 (tile_dim2 = 0)
    g1[5] = 128;                  // tensor_dim0_stride[31:0]
    g1[6] = 0;
    g1[7] = 0;
    i32x4 g2 = {};                // 2-D tensor: groups 2/3 unused (zero)
    i32x4 g3 = {};
    i32x8 g4 = {};                // 6-arg toolchain form: extra zero group
    __builtin_amdgcn_tensor_load_to_lds(g0, g1, g2, g3, g4, 0);
    __builtin_amdgcn_s_wait_tensorcnt(0);
  }
#else
  for (int ch = threadIdx.x; ch < COUT * 16; ch += 256) {  // fallback: v8h copies
    int o = ch >> 4, cc = (ch & 15) * 8;
    *(v8h*)&Wl[o * WLDS_STRIDE + cc] = *(const v8h*)&Wh[o * 128 + cc];
  }
#endif
  if (threadIdx.x < COUT) { sSum[threadIdx.x] = 0.f; sSq[threadIdx.x] = 0.f; }
  __syncthreads();

  const int lane = threadIdx.x & 31;
  const int l16 = lane & 15, hi = lane >> 4;
  const int b = blockIdx.y;
  const int n = blockIdx.x * 8 + (threadIdx.x >> 5);

  const _Float16* ctr = xt + ((size_t)b * NPTS + n) * FDIM;
  int nb = idx[((size_t)b * NPTS + n) * KNN + l16];
  const _Float16* nbr = xt + ((size_t)b * NPTS + nb) * FDIM;

  // feat row l16: c<64 -> nbr-ctr ; c>=64 -> ctr. Chunks never straddle c=64.
  v8h c0 = *(const v8h*)(ctr + hi * 8);
  v8h c1 = *(const v8h*)(ctr + 16 + hi * 8);
  v8h c2 = *(const v8h*)(ctr + 32 + hi * 8);
  v8h c3 = *(const v8h*)(ctr + 48 + hi * 8);
  v8h d0 = *(const v8h*)(nbr + hi * 8) - c0;
  v8h d1 = *(const v8h*)(nbr + 16 + hi * 8) - c1;
  v8h d2 = *(const v8h*)(nbr + 32 + hi * 8) - c2;
  v8h d3 = *(const v8h*)(nbr + 48 + hi * 8) - c3;
  v16h A0 = cat8(d0, d1), A1 = cat8(d2, d3), A2 = cat8(c0, c1), A3 = cat8(c2, c3);

  float* hrow = h + (size_t)(b * NPTS + n) * KNN * COUT;

#pragma unroll
  for (int t = 0; t < 8; ++t) {  // 8 output tiles of 16 channels
    const _Float16* brow = &Wl[(t * 16 + l16) * WLDS_STRIDE];
    v16h B0 = cat8(*(const v8h*)(brow + hi * 8), *(const v8h*)(brow + 16 + hi * 8));
    v16h B1 = cat8(*(const v8h*)(brow + 32 + hi * 8), *(const v8h*)(brow + 48 + hi * 8));
    v16h B2 = cat8(*(const v8h*)(brow + 64 + hi * 8), *(const v8h*)(brow + 80 + hi * 8));
    v16h B3 = cat8(*(const v8h*)(brow + 96 + hi * 8), *(const v8h*)(brow + 112 + hi * 8));
    v8f acc = {};
    acc = wmma_f16(A0, B0, acc);
    acc = wmma_f16(A1, B1, acc);
    acc = wmma_f16(A2, B2, acc);
    acc = wmma_f16(A3, B3, acc);

    float s = 0.f, sq = 0.f;
#pragma unroll
    for (int r = 0; r < 8; ++r) {
      float v = acc[r];
      int kk = r + hi * 8;
      hrow[(size_t)kk * COUT + t * 16 + l16] = v;
      s += v;
      sq += v * v;
    }
    atomicAdd(&sSum[t * 16 + l16], s);  // ds_add_f32
    atomicAdd(&sSq[t * 16 + l16], sq);
  }
  __syncthreads();
  if (threadIdx.x < COUT) {
    atomicAdd(&gSum[threadIdx.x], sSum[threadIdx.x]);
    atomicAdd(&gSq[threadIdx.x], sSq[threadIdx.x]);
  }
}

// ---- Stage 4: fold BN stats into scale/shift -------------------------------
__global__ void k_finalize(const float* __restrict__ gSum, const float* __restrict__ gSq,
                           const float* __restrict__ gamma, const float* __restrict__ beta,
                           float* __restrict__ scale, float* __restrict__ shift) {
  int o = threadIdx.x;
  float mean = gSum[o] * (1.0f / NTOT);
  float var = gSq[o] * (1.0f / NTOT) - mean * mean;
  float sc = gamma[o] * rsqrtf(var + BN_EPS);
  scale[o] = sc;
  shift[o] = beta[o] - mean * sc;
}

// ---- Stage 5: normalize + LeakyReLU + max over K + transposed store --------
__global__ void __launch_bounds__(256) k_out(const float* __restrict__ h,
                                             const float* __restrict__ scale,
                                             const float* __restrict__ shift,
                                             float* __restrict__ out) {
  __shared__ float tile[32 * 129];
  const int b = blockIdx.y;
  const int n0 = blockIdx.x * 32;
  for (int w = threadIdx.x; w < 32 * COUT; w += 256) {
    int o = w & 127, nl = w >> 7;  // o fastest -> coalesced h reads
    const float* hp = h + (size_t)((b * NPTS + n0 + nl) * KNN) * COUT + o;
    __builtin_prefetch(hp + 2 * KNN * COUT, 0, 0);  // global_prefetch_b8: next rows
    float sc = scale[o], sh = shift[o];
    float m = -3.0e38f;
#pragma unroll
    for (int k = 0; k < KNN; ++k) {
      float v = hp[k * COUT] * sc + sh;
      v = (v >= 0.f) ? v : NEG_SLOPE * v;
      m = fmaxf(m, v);
    }
    tile[nl * 129 + o] = m;
  }
  __syncthreads();
  for (int w = threadIdx.x; w < 32 * COUT; w += 256) {
    int o = w >> 5, nl = w & 31;  // n fastest -> coalesced out writes
    out[((size_t)b * COUT + o) * NPTS + n0 + nl] = tile[nl * 129 + o];
  }
}

extern "C" void kernel_launch(void* const* d_in, const int* in_sizes, int n_in,
                              void* d_out, int out_size, void* d_ws, size_t ws_size,
                              hipStream_t stream) {
  const float* x = (const float*)d_in[0];
  const float* W = (const float*)d_in[1];
  const float* gamma = (const float*)d_in[2];
  const float* beta = (const float*)d_in[3];
  float* out = (float*)d_out;
  (void)in_sizes; (void)n_in; (void)out_size; (void)ws_size;

  char* ws = (char*)d_ws;
  size_t off = 0;
  auto salloc = [&](size_t bytes) -> void* {
    void* p = ws + off;
    off = (off + bytes + 255) & ~(size_t)255;
    return p;
  };
  _Float16* xt = (_Float16*)salloc((size_t)BATCH * NPTS * FDIM * sizeof(_Float16));  // 4 MB
  _Float16* Wh = (_Float16*)salloc((size_t)COUT * CIN * sizeof(_Float16));           // 32 KB
  float* xx = (float*)salloc((size_t)BATCH * NPTS * sizeof(float));                  // 128 KB
  int* idx = (int*)salloc((size_t)BATCH * NPTS * KNN * sizeof(int));                 // 2 MB
  float* gSum = (float*)salloc(COUT * sizeof(float));
  float* gSq = (float*)salloc(COUT * sizeof(float));
  float* scale = (float*)salloc(COUT * sizeof(float));
  float* shift = (float*)salloc(COUT * sizeof(float));
  float* dist = (float*)salloc((size_t)NPTS * NPTS * sizeof(float));                 // 64 MB (reused per batch)
  float* h = (float*)salloc((size_t)NTOT * COUT * sizeof(float));                    // 268 MB

  k_prep_x<<<dim3(NPTS / 256, BATCH), 256, 0, stream>>>(x, xt, xx);
  k_prep_w<<<64, 256, 0, stream>>>(W, Wh, gSum, gSq);
  for (int b = 0; b < BATCH; ++b) {  // stream-ordered: dist buffer reuse is safe
    k_gram<<<dim3(NPTS / 16, NPTS / 128), 256, 0, stream>>>(xt, xx, dist, b);
    k_topk<<<NPTS / 8, 256, 0, stream>>>(dist, idx, b);
  }
  k_edgeconv<<<dim3(NPTS / 8, BATCH), 256, 0, stream>>>(xt, Wh, idx, h, gSum, gSq);
  k_finalize<<<1, COUT, 0, stream>>>(gSum, gSq, gamma, beta, scale, shift);
  k_out<<<dim3(NPTS / 32, BATCH), 256, 0, stream>>>(h, scale, shift, out);
}
